// GraphAttentionLayer_9981503996496
// MI455X (gfx1250) — matrix operand order repaired and checked by reading, not compiled, not run on hardware
//
#include <hip/hip_runtime.h>

#define N_NODES 8192
#define IN_F    512
#define OUT_F   256
#define TS      40   // h-tile LDS row stride in halfs: 32 + 8 pad (conflict-free b128)
#define ATS     36   // adj-tile LDS row stride in ints: 32 + 4 pad (TDM pad feature)

typedef __attribute__((ext_vector_type(16))) _Float16 v16h;
typedef __attribute__((ext_vector_type(8)))  _Float16 v8h;
typedef __attribute__((ext_vector_type(8)))  float    v8f;
typedef __attribute__((ext_vector_type(4)))  float    v4f;
typedef __attribute__((ext_vector_type(4)))  int      v4i;
typedef __attribute__((ext_vector_type(4)))  unsigned u32x4;
typedef __attribute__((ext_vector_type(8)))  int      i32x8;
typedef __attribute__((ext_vector_type(4)))  int      i32x4;

union AFrag { v16h v; v8h h8[2]; _Float16 h[16]; };

__device__ __forceinline__ unsigned lds_addr_of(const void* p) {
    // generic LDS pointer = {SHARED_BASE aperture, lds offset} -> low 32 bits
    return (unsigned)(uintptr_t)p;
}

// TDM: DMA a 64-row x 32-int adj tile (row stride 8192 ints) into LDS with
// 4-dword padding every 32 dwords (-> 36-int LDS rows, bank-conflict free).
__device__ __forceinline__ void tdm_load_adj(const int* gtile, unsigned lds_byte)
{
    const unsigned long long ga = (unsigned long long)(uintptr_t)gtile;
    u32x4 g0;
    g0[0] = 1u;                                            // count=1 (user mode)
    g0[1] = lds_byte;                                      // lds_addr
    g0[2] = (unsigned)(ga & 0xffffffffu);                  // global_addr[31:0]
    g0[3] = (unsigned)((ga >> 32) & 0x01ffffffu) | (2u << 30);  // addr[56:32] | type=2

    const unsigned td0 = N_NODES, td1 = N_NODES;           // tensor dims (elements)
    const unsigned tile0 = 32, tile1 = 64;                 // tile dims
    const unsigned long long s0 = N_NODES;                 // dim0 stride (elements)
    i32x8 g1;
    g1[0] = (int)((2u << 16)       // data_size = 4B
                | (1u << 20)       // pad_enable
                | (4u << 22)       // pad_interval: 32 dwords
                | (3u << 25));     // pad_amount:   4 dwords
    g1[1] = (int)((td0 & 0xffffu) << 16);                  // tensor_dim0 lo16
    g1[2] = (int)((td0 >> 16) | ((td1 & 0xffffu) << 16));  // dim0 hi16 | dim1 lo16
    g1[3] = (int)((td1 >> 16) | (tile0 << 16));            // dim1 hi16 | tile_dim0
    g1[4] = (int)tile1;                                    // tile_dim1, tile_dim2=0
    g1[5] = (int)(unsigned)(s0 & 0xffffffffu);             // dim0_stride lo32
    g1[6] = (int)(unsigned)(s0 >> 32);                     // dim0_stride hi16, dim1_stride lo16=0
    g1[7] = 0;
    const i32x4 z4 = {0, 0, 0, 0};
    const i32x8 z8 = {0, 0, 0, 0, 0, 0, 0, 0};
    __builtin_amdgcn_tensor_load_to_lds(g0, g1, z4, z4, z8, 0);
}

// ---------------------------------------------------------------------------
// Phase 1: h = x @ W^T -> hT stored f16 as [OUT_F][N] (transposed for phase 3)
// ---------------------------------------------------------------------------
__global__ __launch_bounds__(256)
void k_h_gemm(const float* __restrict__ x, const float* __restrict__ W,
              _Float16* __restrict__ hT)
{
    __shared__ _Float16 tile[256 * TS];          // W[:, kb..kb+32) as f16
    const int tid  = threadIdx.x;
    const int lane = tid & 31;
    const int wave = tid >> 5;
    const int m    = lane & 15;
    const int ha   = lane >> 4;
    const int m0w  = blockIdx.x * 64 + (wave >> 1) * 16;
    const int ntB  = (wave & 1) * 8;

    v8f acc[8];
#pragma unroll
    for (int i = 0; i < 8; ++i)
#pragma unroll
        for (int v = 0; v < 8; ++v) acc[i][v] = 0.f;

    const float* xr = x + (size_t)(m0w + m) * IN_F;
    const float* wr = W + (size_t)tid * IN_F;

    for (int kb = 0; kb < IN_F; kb += 32) {
        __syncthreads();
#pragma unroll
        for (int i = 0; i < 4; ++i) {            // stage W slice, fused f32->f16
            v4f c0 = *(const v4f*)(wr + kb + i * 8);
            v4f c1 = *(const v4f*)(wr + kb + i * 8 + 4);
            v8h hs;
#pragma unroll
            for (int q = 0; q < 4; ++q) { hs[q] = (_Float16)c0[q]; hs[q + 4] = (_Float16)c1[q]; }
            *(v8h*)&tile[tid * TS + i * 8] = hs;
        }
        __syncthreads();

        AFrag a;                                  // 16x32 f16 A layout
        {
            v4f c0 = *(const v4f*)(xr + kb + 8 * ha);
            v4f c1 = *(const v4f*)(xr + kb + 8 * ha + 4);
            v4f c2 = *(const v4f*)(xr + kb + 16 + 8 * ha);
            v4f c3 = *(const v4f*)(xr + kb + 16 + 8 * ha + 4);
#pragma unroll
            for (int q = 0; q < 4; ++q) {
                a.h[q]      = (_Float16)c0[q];  a.h[q + 4]  = (_Float16)c1[q];
                a.h[q + 8]  = (_Float16)c2[q];  a.h[q + 12] = (_Float16)c3[q];
            }
        }
#pragma unroll
        for (int nt = 0; nt < 8; ++nt) {
            const int fr = (ntB + nt) * 16 + m;
            AFrag b;
            b.h8[0] = *(const v8h*)&tile[fr * TS + 8 * ha];
            b.h8[1] = *(const v8h*)&tile[fr * TS + 16 + 8 * ha];
            acc[nt] = __builtin_amdgcn_wmma_f32_16x16x32_f16(
                false, a.v, false, b.v, (short)0, acc[nt], false, false);
        }
    }

    const int node0 = m0w + 8 * ha;
#pragma unroll
    for (int nt = 0; nt < 8; ++nt) {
        const int f = (ntB + nt) * 16 + m;
        v8h hv;
#pragma unroll
        for (int v = 0; v < 8; ++v) hv[v] = (_Float16)acc[nt][v];
        *(v8h*)(hT + (size_t)f * N_NODES + node0) = hv;
    }
}

// ---------------------------------------------------------------------------
// Phase 2: s_i = h@a[:F], s_j = h@a[F:]
// ---------------------------------------------------------------------------
__global__ __launch_bounds__(256)
void k_scores(const _Float16* __restrict__ hT, const float* __restrict__ a,
              float* __restrict__ s_i, float* __restrict__ s_j)
{
    const int n = blockIdx.x * 256 + threadIdx.x;
    float si = 0.f, sj = 0.f;
#pragma unroll 4
    for (int f = 0; f < OUT_F; ++f) {
        const float hv = (float)hT[(size_t)f * N_NODES + n];
        si += hv * a[f];
        sj += hv * a[OUT_F + f];
    }
    s_i[n] = si; s_j[n] = sj;
}

// ---------------------------------------------------------------------------
// Phase 3: fused mask + online softmax + attention@h + ELU, double-buffered:
//   hT tile  -> LDS via global_load_async_to_lds_b128 (ASYNCcnt)
//   adj tile -> LDS via TDM tensor_load_to_lds        (TENSORcnt, wave 0)
// ---------------------------------------------------------------------------
__global__ __launch_bounds__(256)
void k_attn(const int* __restrict__ adj, const _Float16* __restrict__ hT,
            const float* __restrict__ s_i, const float* __restrict__ s_j,
            float* __restrict__ out)
{
    __shared__ _Float16 tile[2][256 * TS];       // 2 x 20 KB
    __shared__ int      adjT[2][64 * ATS];       // 2 x  9 KB (TDM-padded rows)

    const int tid  = threadIdx.x;
    const int lane = tid & 31;
    const int wave = tid >> 5;
    const int m    = lane & 15;
    const int ha   = lane >> 4;
    const int m0w  = blockIdx.x * 64 + (wave >> 1) * 16;
    const int ntB  = (wave & 1) * 8;
    const int rAdj = (wave >> 1) * 16 + m;       // row within the 64-row adj tile

    v8f acc[8];
#pragma unroll
    for (int i = 0; i < 8; ++i)
#pragma unroll
        for (int v = 0; v < 8; ++v) acc[i][v] = 0.f;

    float rm = -3e38f, lsum = 0.f;
    const float si       = s_i[m0w + m];
    const int*  adjBlk   = adj + (size_t)(blockIdx.x * 64) * N_NODES;
    const _Float16* hrow = hT + (size_t)tid * N_NODES;   // staging: thread t = feature row t

    // async-stage one hT column-tile into LDS buffer `buf`
    auto issue_h = [&](int jb, int buf) {
#pragma unroll
        for (int i = 0; i < 4; ++i) {
            const unsigned la = lds_addr_of(&tile[buf][tid * TS + i * 8]);
            const unsigned long long ga = (unsigned long long)(uintptr_t)(hrow + jb + i * 8);
            asm volatile("global_load_async_to_lds_b128 %0, %1, off"
                         :: "v"(la), "v"(ga) : "memory");
        }
    };

    // prologue: stage jb = 0 into buffer 0
    if (wave == 0) tdm_load_adj(adjBlk + 0, lds_addr_of(&adjT[0][0]));
    issue_h(0, 0);
    int cur = 0;

    for (int jb = 0; jb < N_NODES; jb += 32) {
        asm volatile("s_wait_asynccnt 0x0" ::: "memory");
        if (wave == 0) __builtin_amdgcn_s_wait_tensorcnt(0);
        __syncthreads();                          // buffer `cur` ready for everyone
        if (jb + 32 < N_NODES) {                  // prefetch next block into cur^1
            if (wave == 0) tdm_load_adj(adjBlk + (jb + 32), lds_addr_of(&adjT[cur ^ 1][0]));
            issue_h(jb + 32, cur ^ 1);
        }

        const int k0 = 8 * ha;
        const int* arow = &adjT[cur][rAdj * ATS];
        v4i a0  = *(const v4i*)(arow + k0);
        v4i a0b = *(const v4i*)(arow + k0 + 4);
        v4i a1  = *(const v4i*)(arow + k0 + 16);
        v4i a1b = *(const v4i*)(arow + k0 + 20);
        v4f sj0  = *(const v4f*)(s_j + jb + k0);
        v4f sj0b = *(const v4f*)(s_j + jb + k0 + 4);
        v4f sj1  = *(const v4f*)(s_j + jb + k0 + 16);
        v4f sj1b = *(const v4f*)(s_j + jb + k0 + 20);

        float e[16]; int msk[16];
#pragma unroll
        for (int q = 0; q < 4; ++q) {
            float t;
            t = si + sj0[q];  e[q]      = t > 0.f ? t : 0.01f * t;  msk[q]      = a0[q];
            t = si + sj0b[q]; e[q + 4]  = t > 0.f ? t : 0.01f * t;  msk[q + 4]  = a0b[q];
            t = si + sj1[q];  e[q + 8]  = t > 0.f ? t : 0.01f * t;  msk[q + 8]  = a1[q];
            t = si + sj1b[q]; e[q + 12] = t > 0.f ? t : 0.01f * t;  msk[q + 12] = a1b[q];
        }

        float bm = -3e38f;
#pragma unroll
        for (int i = 0; i < 16; ++i) if (msk[i] > 0) bm = fmaxf(bm, e[i]);
        bm = fmaxf(bm, __shfl_xor(bm, 16));
        const float rmn   = fmaxf(rm, bm);
        const float alpha = __expf(rm - rmn);
        rm = rmn;

        AFrag p;
        float bs = 0.f;
#pragma unroll
        for (int i = 0; i < 16; ++i) {
            const float pv = (msk[i] > 0) ? __expf(e[i] - rm) : 0.f;
            bs += pv;
            p.h[i] = (_Float16)pv;
        }
        bs += __shfl_xor(bs, 16);
        lsum = lsum * alpha + bs;

#pragma unroll
        for (int v = 0; v < 8; ++v) {
            const float av = __shfl(alpha, v + ((lane & 16) >> 1));
#pragma unroll
            for (int nt = 0; nt < 8; ++nt) acc[nt][v] *= av;
        }

#pragma unroll
        for (int nt = 0; nt < 8; ++nt) {
            const int fr = (ntB + nt) * 16 + m;
            AFrag b;
            b.h8[0] = *(const v8h*)&tile[cur][fr * TS + 8 * ha];
            b.h8[1] = *(const v8h*)&tile[cur][fr * TS + 16 + 8 * ha];
            acc[nt] = __builtin_amdgcn_wmma_f32_16x16x32_f16(
                false, p.v, false, b.v, (short)0, acc[nt], false, false);
        }
        cur ^= 1;
    }

    const float linv = 1.f / fmaxf(lsum, 1e-30f);
#pragma unroll
    for (int nt = 0; nt < 8; ++nt) {
        const int f = (ntB + nt) * 16 + m;
#pragma unroll
        for (int v = 0; v < 8; ++v) {
            const float lv = __shfl(linv, v + ((lane & 16) >> 1));
            float o = acc[nt][v] * lv;
            o = (o > 0.f) ? o : (__expf(o) - 1.f);
            out[(size_t)(m0w + v + 8 * ha) * OUT_F + f] = o;
        }
    }
}

// ---------------------------------------------------------------------------
extern "C" void kernel_launch(void* const* d_in, const int* in_sizes, int n_in,
                              void* d_out, int out_size, void* d_ws, size_t ws_size,
                              hipStream_t stream)
{
    const float* x   = (const float*)d_in[0];
    const int*   adj = (const int*)d_in[1];
    const float* W   = (const float*)d_in[2];
    const float* a   = (const float*)d_in[3];
    float* out = (float*)d_out;

    char* ws = (char*)d_ws;
    _Float16* hT = (_Float16*)ws;                                  // 4 MB
    float* s_i   = (float*)(ws + (size_t)OUT_F * N_NODES * 2);     // 32 KB
    float* s_j   = s_i + N_NODES;                                  // 32 KB

    hipLaunchKernelGGL(k_h_gemm, dim3(N_NODES / 64),  dim3(256), 0, stream, x, W, hT);
    hipLaunchKernelGGL(k_scores, dim3(N_NODES / 256), dim3(256), 0, stream, hT, a, s_i, s_j);
    hipLaunchKernelGGL(k_attn,   dim3(N_NODES / 64),  dim3(256), 0, stream, adj, hT, s_i, s_j, out);
}
